// BertSelfAttention_58755152609881
// MI455X (gfx1250) — matrix operand order repaired and checked by reading, not compiled, not run on hardware
//
#include <hip/hip_runtime.h>
#include <hip/hip_bf16.h>

#define B_  2
#define S_  2048
#define H_  1024
#define NH  16
#define HD  64

typedef __attribute__((ext_vector_type(16))) __bf16 v16bf;
typedef __attribute__((ext_vector_type(8)))  __bf16 v8bf;
typedef __attribute__((ext_vector_type(4)))  __bf16 v4bf;
typedef __attribute__((ext_vector_type(8)))  float  v8f;
typedef __attribute__((ext_vector_type(4)))  int    i32x4;

__device__ __forceinline__ __bf16 f2bf(float x) { return (__bf16)x; }
__device__ __forceinline__ v16bf cat16(v8bf lo, v8bf hi) {
  return __builtin_shufflevector(lo, hi, 0,1,2,3,4,5,6,7,8,9,10,11,12,13,14,15);
}

// ---- gfx1250 async global->LDS copy (ASYNCcnt path), with sync fallback ----
#if defined(__has_builtin)
#if __has_builtin(__builtin_amdgcn_global_load_async_to_lds_b128) && \
    __has_builtin(__builtin_amdgcn_s_wait_asynccnt)
#define HAVE_ASYNC_LDS 1
#endif
#endif
#ifndef HAVE_ASYNC_LDS
#define HAVE_ASYNC_LDS 0
#endif

#if HAVE_ASYNC_LDS
typedef __attribute__((address_space(1))) i32x4 g_i32x4;
typedef __attribute__((address_space(3))) i32x4 l_i32x4;
#endif

__device__ __forceinline__ void copy16_to_lds(const __bf16* g, __bf16* l) {
#if HAVE_ASYNC_LDS
  __builtin_amdgcn_global_load_async_to_lds_b128(
      (g_i32x4*)g, (l_i32x4*)l, 0, 0);
#else
  *reinterpret_cast<v8bf*>(l) = *reinterpret_cast<const v8bf*>(g);
#endif
}
__device__ __forceinline__ void drain_async() {
#if HAVE_ASYNC_LDS
  __builtin_amdgcn_s_wait_asynccnt(0);
#endif
}

// ---------------------------------------------------------------------------
// Kernel 1: gate g = sigmoid(hs @ Wg + bg) and powterm = (S-2)^(1-g)
// ---------------------------------------------------------------------------
__global__ __launch_bounds__(256)
void gate_kernel(const float* __restrict__ hs, const float* __restrict__ Wg,
                 const float* __restrict__ bg, float* __restrict__ gate,
                 float* __restrict__ powg) {
  int lane = threadIdx.x & 31;
  int wid  = threadIdx.x >> 5;
  int row  = blockIdx.x * 8 + wid;
  const float* hrow = hs + (long)row * H_;
  float acc = 0.f;
  #pragma unroll 8
  for (int i = lane; i < H_; i += 32) acc += hrow[i] * Wg[i];
  #pragma unroll
  for (int off = 16; off >= 1; off >>= 1) acc += __shfl_xor(acc, off, 32);
  if (lane == 0) {
    float g = 1.f / (1.f + __expf(-(acc + bg[0])));
    gate[row] = g;
    powg[row] = __expf((1.f - g) * __logf((float)(S_ - 2)));
  }
}

// ---------------------------------------------------------------------------
// Kernel 2: log_pen[b,i,j] (shared by all 16 heads -> precompute once)
// ---------------------------------------------------------------------------
__global__ __launch_bounds__(256)
void pen_kernel(const float* __restrict__ gate, const float* __restrict__ powg,
                float* __restrict__ pen) {
  int idx = blockIdx.x * 256 + threadIdx.x;
  int j = idx & (S_ - 1);
  int t = idx >> 11;
  int i = t & (S_ - 1);
  int b = t >> 11;
  float zi = gate[b * S_ + i];
  float zj = gate[b * S_ + j];
  float pw = powg[b * S_ + i];
  float res = (1.f - zi) * fmaxf(1.f - zi - zj, 0.f)
            + zi * fminf(1.f - zi + zj, 1.f);
  float absij = fabsf((float)(i - j));
  float scope = fminf(fmaxf(pw + 2.f - absij, 1e-32f), 1.f);
  pen[idx] = __logf(fmaxf(res * scope, 1e-32f));
}

// ---------------------------------------------------------------------------
// Kernel 2b: X fp32 -> bf16 (row-major, unchanged layout)
// ---------------------------------------------------------------------------
__global__ __launch_bounds__(256)
void cvt_x(const float* __restrict__ X, __bf16* __restrict__ Xb) {
  long i = (long)(blockIdx.x * 256 + threadIdx.x) * 4;
  float4 v = *reinterpret_cast<const float4*>(X + i);
  v4bf p = { f2bf(v.x), f2bf(v.y), f2bf(v.z), f2bf(v.w) };
  *reinterpret_cast<v4bf*>(Xb + i) = p;
}

// ---------------------------------------------------------------------------
// Kernel 2c: W fp32 [k][n] -> bf16 transposed Wt [n][k] via LDS 32x32 tiles
// ---------------------------------------------------------------------------
__global__ __launch_bounds__(256)
void transpose_w(const float* __restrict__ W, __bf16* __restrict__ Wt) {
  __shared__ float t[32][33];
  int k0 = blockIdx.x * 32;
  int n0 = blockIdx.y * 32;
  int tx = threadIdx.x & 31, ty = threadIdx.x >> 5;   // 32 x 8
  #pragma unroll
  for (int j = 0; j < 4; ++j)
    t[ty + j * 8][tx] = W[(long)(k0 + ty + j * 8) * H_ + n0 + tx];
  __syncthreads();
  #pragma unroll
  for (int j = 0; j < 4; ++j) {
    int n = ty + j * 8;
    Wt[(long)(n0 + n) * H_ + k0 + tx] = f2bf(t[tx][n]);
  }
}

// ---------------------------------------------------------------------------
// Kernel 3: Xb[4096,1024]bf16 @ Wt[n][k]bf16 + bias -> bf16 head-split.
// Double-buffered async-LDS tiles; pure async-copy + WMMA K-loop.
// vtrans=0: out[bh][s][d] (Q,K);  vtrans=1: out[bh][d][s] (V pre-transposed)
// ---------------------------------------------------------------------------
__global__ __launch_bounds__(128)
void qkv_gemm(const __bf16* __restrict__ Xb, const __bf16* __restrict__ Wt,
              const float* __restrict__ bias, __bf16* __restrict__ out,
              int vtrans) {
  __shared__ __bf16 As[2][64][72];    // [m][k]
  __shared__ __bf16 Bs[2][64][72];    // [n][k]
  int tid  = threadIdx.x;
  int lane = tid & 31, wid = tid >> 5;
  int l16  = lane & 15, hf = lane >> 4;
  int row0 = blockIdx.x * 64;
  int n0   = blockIdx.y * 64;

  auto issue_tile = [&](int k0, int buf) {
    #pragma unroll
    for (int it = 0; it < 4; ++it) {               // 512 16B chunks per tile
      int e = it * 128 + tid;
      int r = e >> 3, c8 = (e & 7) << 3;
      copy16_to_lds(Xb + (long)(row0 + r) * H_ + k0 + c8, &As[buf][r][c8]);
      copy16_to_lds(Wt + (long)(n0 + r) * H_ + k0 + c8, &Bs[buf][r][c8]);
    }
  };

  v8f c[4] = {};
  issue_tile(0, 0);
  int cur = 0;
  for (int k0 = 0; k0 < H_; k0 += 64) {
    drain_async();
    __syncthreads();
    if (k0 + 64 < H_) issue_tile(k0 + 64, cur ^ 1);
    int wm0 = wid * 16;
    #pragma unroll
    for (int kk = 0; kk < 64; kk += 32) {
      v16bf a = cat16(
          *reinterpret_cast<const v8bf*>(&As[cur][wm0 + l16][kk + hf * 8]),
          *reinterpret_cast<const v8bf*>(&As[cur][wm0 + l16][kk + 16 + hf * 8]));
      #pragma unroll
      for (int ns = 0; ns < 4; ++ns) {
        v16bf bfrag = cat16(
            *reinterpret_cast<const v8bf*>(&Bs[cur][ns * 16 + l16][kk + hf * 8]),
            *reinterpret_cast<const v8bf*>(&Bs[cur][ns * 16 + l16][kk + 16 + hf * 8]));
        c[ns] = __builtin_amdgcn_wmma_f32_16x16x32_bf16(
            false, a, false, bfrag, (short)0, c[ns], false, false);
      }
    }
    cur ^= 1;
  }
  #pragma unroll
  for (int ns = 0; ns < 4; ++ns) {
    int col = n0 + ns * 16 + l16;
    float bias_v = bias[col];
    int head = col >> 6, d = col & 63;
    #pragma unroll
    for (int r = 0; r < 8; ++r) {
      int row = row0 + wid * 16 + hf * 8 + r;
      int bb = row >> 11, s = row & (S_ - 1);
      long oidx = vtrans ? (((long)(bb * NH + head)) * HD + d) * S_ + s
                         : (((long)(bb * NH + head)) * S_ + s) * HD + d;
      out[oidx] = f2bf(c[ns][r] + bias_v);
    }
  }
}

// ---------------------------------------------------------------------------
// Kernel 4: flash attention with precomputed log_pen bias.
// Q,K: [bh][s][d] bf16; V: [bh][d][s] bf16. Double-buffered async K/V tiles,
// one online-softmax update per 64-token block.
// ---------------------------------------------------------------------------
__global__ __launch_bounds__(128)
void attn_kernel(const __bf16* __restrict__ Q, const __bf16* __restrict__ K,
                 const __bf16* __restrict__ V, const float* __restrict__ pen,
                 float* __restrict__ out) {
  __shared__ __bf16 Ks[2][64][72];                 // [token][d]
  __shared__ __bf16 Vt[2][64][72];                 // [d][token]
  __shared__ __bf16 Ps[4][16][72];                 // per-wave P (16 x 64)
  int tid  = threadIdx.x;
  int lane = tid & 31, wid = tid >> 5;
  int l16  = lane & 15, hf = lane >> 4;
  int bh = blockIdx.y;
  int b = bh >> 4, h = bh & 15;
  int q0 = blockIdx.x * 64;
  int qrow0 = q0 + wid * 16;

  const __bf16* Qh = Q + (long)bh * S_ * HD;
  const __bf16* Kh = K + (long)bh * S_ * HD;
  const __bf16* Vh = V + (long)bh * HD * S_;
  const float* penrow = pen + ((long)b * S_ + qrow0) * S_;

  auto issue_tile = [&](int kb, int buf) {
    #pragma unroll
    for (int it = 0; it < 4; ++it) {               // 512 16B chunks per tile
      int e = it * 128 + tid;
      int r = e >> 3, c8 = (e & 7) << 3;
      copy16_to_lds(Kh + (long)(kb + r) * HD + c8, &Ks[buf][r][c8]);
      copy16_to_lds(Vh + (long)r * S_ + kb + c8, &Vt[buf][r][c8]);
    }
  };

  // Q fragments, resident all kernel (A-layout: two 16B chunks each)
  const __bf16* qbase = Qh + (long)(qrow0 + l16) * HD;
  v16bf aq0 = cat16(*reinterpret_cast<const v8bf*>(qbase + hf * 8),
                    *reinterpret_cast<const v8bf*>(qbase + 16 + hf * 8));
  v16bf aq1 = cat16(*reinterpret_cast<const v8bf*>(qbase + 32 + hf * 8),
                    *reinterpret_cast<const v8bf*>(qbase + 48 + hf * 8));

  v8f o[4] = {};
  float row_m[8], row_l[8];
  #pragma unroll
  for (int r = 0; r < 8; ++r) { row_m[r] = -1e30f; row_l[r] = 0.f; }

  issue_tile(0, 0);
  int cur = 0;
  for (int kb = 0; kb < S_; kb += 64) {
    drain_async();
    __syncthreads();
    if (kb + 64 < S_) issue_tile(kb + 64, cur ^ 1);

    // ---- scores for all 64 columns (4 x 16-col fragments) ----
    float sv[4][8];
    #pragma unroll
    for (int t4 = 0; t4 < 4; ++t4) {
      int kt = t4 * 16;
      v16bf b0 = cat16(
          *reinterpret_cast<const v8bf*>(&Ks[cur][kt + l16][hf * 8]),
          *reinterpret_cast<const v8bf*>(&Ks[cur][kt + l16][16 + hf * 8]));
      v16bf b1 = cat16(
          *reinterpret_cast<const v8bf*>(&Ks[cur][kt + l16][32 + hf * 8]),
          *reinterpret_cast<const v8bf*>(&Ks[cur][kt + l16][48 + hf * 8]));
      v8f cs = {};
      cs = __builtin_amdgcn_wmma_f32_16x16x32_bf16(false, aq0, false, b0,
                                                   (short)0, cs, false, false);
      cs = __builtin_amdgcn_wmma_f32_16x16x32_bf16(false, aq1, false, b1,
                                                   (short)0, cs, false, false);
      #pragma unroll
      for (int r = 0; r < 8; ++r) {
        int mr = hf * 8 + r;
        float pb = penrow[(long)mr * S_ + kb + kt + l16];
        sv[t4][r] = cs[r] * 0.125f + pb;
      }
    }
    // ---- single online-softmax update over 64 columns ----
    #pragma unroll
    for (int r = 0; r < 8; ++r) {
      float mx = fmaxf(fmaxf(sv[0][r], sv[1][r]), fmaxf(sv[2][r], sv[3][r]));
      #pragma unroll
      for (int off = 8; off >= 1; off >>= 1) mx = fmaxf(mx, __shfl_xor(mx, off, 32));
      float nm = fmaxf(row_m[r], mx);
      float corr = __expf(row_m[r] - nm);
      float p0 = __expf(sv[0][r] - nm);
      float p1 = __expf(sv[1][r] - nm);
      float p2 = __expf(sv[2][r] - nm);
      float p3 = __expf(sv[3][r] - nm);
      float sum = (p0 + p1) + (p2 + p3);
      #pragma unroll
      for (int off = 8; off >= 1; off >>= 1) sum += __shfl_xor(sum, off, 32);
      row_l[r] = row_l[r] * corr + sum;
      row_m[r] = nm;
      sv[0][r] = p0; sv[1][r] = p1; sv[2][r] = p2; sv[3][r] = p3;
      #pragma unroll
      for (int f = 0; f < 4; ++f) o[f][r] *= corr;
    }
    // ---- bounce P through LDS: C layout -> A layout ----
    #pragma unroll
    for (int t4 = 0; t4 < 4; ++t4)
      #pragma unroll
      for (int r = 0; r < 8; ++r)
        Ps[wid][hf * 8 + r][t4 * 16 + l16] = f2bf(sv[t4][r]);
    __builtin_amdgcn_fence(__ATOMIC_ACQ_REL, "wavefront");
    // ---- PV: two K=32 steps ----
    #pragma unroll
    for (int sub = 0; sub < 2; ++sub) {
      v16bf ap = cat16(
          *reinterpret_cast<const v8bf*>(&Ps[wid][l16][sub * 32 + hf * 8]),
          *reinterpret_cast<const v8bf*>(&Ps[wid][l16][sub * 32 + 16 + hf * 8]));
      #pragma unroll
      for (int dt = 0; dt < 4; ++dt) {
        v16bf bv = cat16(
            *reinterpret_cast<const v8bf*>(&Vt[cur][dt * 16 + l16][sub * 32 + hf * 8]),
            *reinterpret_cast<const v8bf*>(&Vt[cur][dt * 16 + l16][sub * 32 + 16 + hf * 8]));
        o[dt] = __builtin_amdgcn_wmma_f32_16x16x32_bf16(
            false, ap, false, bv, (short)0, o[dt], false, false);
      }
    }
    cur ^= 1;
  }
  #pragma unroll
  for (int r = 0; r < 8; ++r) {
    int mr = hf * 8 + r;
    float inv = 1.0f / row_l[r];
    int s = qrow0 + mr;
    #pragma unroll
    for (int dt = 0; dt < 4; ++dt)
      out[((long)b * S_ + s) * H_ + h * HD + dt * 16 + l16] = o[dt][r] * inv;
  }
}

// ---------------------------------------------------------------------------
extern "C" void kernel_launch(void* const* d_in, const int* in_sizes, int n_in,
                              void* d_out, int out_size, void* d_ws, size_t ws_size,
                              hipStream_t stream) {
  const float* hs = (const float*)d_in[0];
  const float* Wq = (const float*)d_in[1];
  const float* bq = (const float*)d_in[2];
  const float* Wk = (const float*)d_in[3];
  const float* bk = (const float*)d_in[4];
  const float* Wv = (const float*)d_in[5];
  const float* bv = (const float*)d_in[6];
  const float* Wg = (const float*)d_in[7];
  const float* bg = (const float*)d_in[8];

  float*  gate = (float*)d_ws;
  float*  powg = gate + B_ * S_;
  float*  pen  = powg + B_ * S_;
  __bf16* q    = (__bf16*)(pen + (long)B_ * S_ * S_);
  __bf16* k    = q + (long)B_ * S_ * H_;
  __bf16* v    = k + (long)B_ * S_ * H_;
  __bf16* xb   = v + (long)B_ * S_ * H_;            // X in bf16
  __bf16* wtq  = xb + (long)B_ * S_ * H_;           // W^T in bf16
  __bf16* wtk  = wtq + (long)H_ * H_;
  __bf16* wtv  = wtk + (long)H_ * H_;

  gate_kernel<<<(B_ * S_) / 8, 256, 0, stream>>>(hs, Wg, bg, gate, powg);
  pen_kernel<<<((long)B_ * S_ * S_) / 256, 256, 0, stream>>>(gate, powg, pen);
  cvt_x<<<((long)B_ * S_ * H_) / 4 / 256, 256, 0, stream>>>(hs, xb);
  transpose_w<<<dim3(H_ / 32, H_ / 32), 256, 0, stream>>>(Wq, wtq);
  transpose_w<<<dim3(H_ / 32, H_ / 32), 256, 0, stream>>>(Wk, wtk);
  transpose_w<<<dim3(H_ / 32, H_ / 32), 256, 0, stream>>>(Wv, wtv);
  qkv_gemm<<<dim3((B_ * S_) / 64, H_ / 64), 128, 0, stream>>>(xb, wtq, bq, q, 0);
  qkv_gemm<<<dim3((B_ * S_) / 64, H_ / 64), 128, 0, stream>>>(xb, wtk, bk, k, 0);
  qkv_gemm<<<dim3((B_ * S_) / 64, H_ / 64), 128, 0, stream>>>(xb, wtv, bv, v, 1);
  attn_kernel<<<dim3(S_ / 64, B_ * NH), 128, 0, stream>>>(q, k, v, pen,
                                                          (float*)d_out);
}
